// BlockCache_29789893165563
// MI455X (gfx1250) — matrix operand order repaired and checked by reading, not compiled, not run on hardware
//
#include <hip/hip_runtime.h>
#include <hip/hip_bf16.h>
#include <math.h>

#define HIDDEN 4096
#define NHEADS 32
#define NKV 2
#define HD 128
#define ROT 64
#define FFN 13696
#define SEQ 4096
#define PAST 4095
#define BATCH 32
#define QKV_O (HIDDEN + 2 * NKV * HD) /* 4608 */
#define ASPLIT 4                      /* attention key splits */
#define PSTRIDE 130                   /* per-(head,split) partial: m, l, acc[128] */
#define GSPLIT 4                      /* GEMM K-splits for small-N GEMMs */

typedef __attribute__((ext_vector_type(16))) __bf16 v16bf;
typedef __attribute__((ext_vector_type(8))) unsigned v8u;
typedef __attribute__((ext_vector_type(4))) unsigned v4u;
typedef __attribute__((ext_vector_type(8))) float v8f;
typedef __attribute__((ext_vector_type(4))) float v4f;

// Pack two f32 into packed bf16 (truncate) with one v_perm_b32:
// result low half = x>>16, high half = y>>16.
__device__ __forceinline__ unsigned pk_bf16(float x, float y) {
    union { float f; unsigned u; } a, b; a.f = x; b.f = y;
    return __builtin_amdgcn_perm(b.u, a.u, 0x07060302u);
}

// ---------------------------------------------------------------------------
// GEMM: C[M=32, N] = A[32, K](bf16) * W[N, K](f32->bf16)^T, via bf16 WMMA.
// One wave owns a 16-column tile and BOTH 16-row M tiles -> each weight
// fragment loaded exactly once (non-temporal: pure stream, keep L2 for KV).
// A is pre-converted bf16 -> fragments are two 16B b128 loads per tile,
// no conversion ops. W converted in-register with v_perm (8 perms/step).
// grid = (N/128, KS). KS>1: raw f32 partials to Cp[ks][32][N];
// KS==1: direct store with bias/resid.
// Per-lane fragment layouts per CDNA5 ISA 7.12.2 (16-bit A 16x32, B 32x16).
// K % (32*KS) == 0, N % 128 == 0. Block = 256 threads = 8 waves.
// ---------------------------------------------------------------------------
__global__ __launch_bounds__(256) void gemm_bf16_wmma(
    const unsigned short* __restrict__ Xb, const float* __restrict__ W,
    float* __restrict__ Cp, const float* __restrict__ bias,
    const float* __restrict__ resid, int K, int N, int KS)
{
    const int lane = threadIdx.x & 31;
    const int wid  = threadIdx.x >> 5;
    const int n0   = blockIdx.x * 128 + wid * 16;
    const int l15  = lane & 15;
    const int hi   = lane >> 4;
    const int klen = K / KS;
    const int kbeg = blockIdx.y * klen;

    const unsigned short* xa0 = Xb + (size_t)l15 * K + hi * 8;        // M rows 0..15
    const unsigned short* xa1 = Xb + (size_t)(16 + l15) * K + hi * 8; // M rows 16..31
    const float*          wb  = W  + (size_t)(n0 + l15) * K + hi * 8;

    v8f acc0 = {}; v8f acc1 = {};

#pragma unroll 2
    for (int k0 = kbeg; k0 < kbeg + klen; k0 += 32) {
        const v4f* bp = (const v4f*)(wb + k0);
        v4f b0 = __builtin_nontemporal_load(bp + 0);
        v4f b1 = __builtin_nontemporal_load(bp + 1);
        v4f b2 = __builtin_nontemporal_load(bp + 4);
        v4f b3 = __builtin_nontemporal_load(bp + 5);
        const v4u* a0p = (const v4u*)(xa0 + k0);
        v4u a0lo = a0p[0], a0hi = a0p[2];     // halves k+0..7 and k+16..23
        const v4u* a1p = (const v4u*)(xa1 + k0);
        v4u a1lo = a1p[0], a1hi = a1p[2];

        v8u a0u, a1u, bu;
#pragma unroll
        for (int e = 0; e < 4; ++e) {
            a0u[e] = a0lo[e]; a0u[4 + e] = a0hi[e];
            a1u[e] = a1lo[e]; a1u[4 + e] = a1hi[e];
        }
        bu[0] = pk_bf16(b0[0], b0[1]); bu[1] = pk_bf16(b0[2], b0[3]);
        bu[2] = pk_bf16(b1[0], b1[1]); bu[3] = pk_bf16(b1[2], b1[3]);
        bu[4] = pk_bf16(b2[0], b2[1]); bu[5] = pk_bf16(b2[2], b2[3]);
        bu[6] = pk_bf16(b3[0], b3[1]); bu[7] = pk_bf16(b3[2], b3[3]);

        v16bf a0 = __builtin_bit_cast(v16bf, a0u);
        v16bf a1 = __builtin_bit_cast(v16bf, a1u);
        v16bf b  = __builtin_bit_cast(v16bf, bu);
        acc0 = __builtin_amdgcn_wmma_f32_16x16x32_bf16(false, a0, false, b, (short)0, acc0, false, false);
        acc1 = __builtin_amdgcn_wmma_f32_16x16x32_bf16(false, a1, false, b, (short)0, acc1, false, false);
    }

    const int n = n0 + l15;
    if (KS == 1) {
        const float bv = bias ? bias[n] : 0.0f;
#pragma unroll
        for (int r = 0; r < 8; ++r) {
            const int m0 = hi * 8 + r;
            float v0 = acc0[r] + bv;
            float v1 = acc1[r] + bv;
            if (resid) {
                v0 += resid[(size_t)m0 * N + n];
                v1 += resid[(size_t)(m0 + 16) * N + n];
            }
            Cp[(size_t)m0 * N + n]        = v0;
            Cp[(size_t)(m0 + 16) * N + n] = v1;
        }
    } else {
        float* cp = Cp + (size_t)blockIdx.y * BATCH * N;
#pragma unroll
        for (int r = 0; r < 8; ++r) {
            const int m0 = hi * 8 + r;
            cp[(size_t)m0 * N + n]        = acc0[r];
            cp[(size_t)(m0 + 16) * N + n] = acc1[r];
        }
    }
}

// ---------------------------------------------------------------------------
// Split-K reduction + bias + residual. total = BATCH*N elements.
// ---------------------------------------------------------------------------
__global__ __launch_bounds__(256) void gemm_reduce_k(
    const float* __restrict__ Cp, float* __restrict__ C,
    const float* __restrict__ bias, const float* __restrict__ resid,
    int N, int KS, int total)
{
    const int idx = blockIdx.x * 256 + threadIdx.x;
    if (idx >= total) return;
    float s = 0.f;
    for (int k = 0; k < KS; ++k) s += Cp[(size_t)k * total + idx];
    if (bias)  s += bias[idx % N];
    if (resid) s += resid[idx];
    C[idx] = s;
}

// ---------------------------------------------------------------------------
// RMSNorm: one block per batch row; emits packed bf16 (GEMM A operand).
// ---------------------------------------------------------------------------
__global__ __launch_bounds__(256) void rmsnorm_k(const float* __restrict__ x,
                                                 const float* __restrict__ w,
                                                 unsigned* __restrict__ ob)
{
    __shared__ float red[256];
    const int b = blockIdx.x;
    const float* xr = x + (size_t)b * HIDDEN;
    float s = 0.f;
    for (int i = threadIdx.x; i < HIDDEN; i += 256) { float v = xr[i]; s += v * v; }
    red[threadIdx.x] = s; __syncthreads();
    for (int st = 128; st > 0; st >>= 1) {
        if (threadIdx.x < st) red[threadIdx.x] += red[threadIdx.x + st];
        __syncthreads();
    }
    const float inv = rsqrtf(red[0] / (float)HIDDEN + 1e-5f);
    const v4f* xp = (const v4f*)xr;
    const v4f* wp = (const v4f*)w;
    for (int i = threadIdx.x; i < HIDDEN / 4; i += 256) {
        v4f xv = xp[i], wv = wp[i];
        ob[(size_t)b * (HIDDEN / 2) + 2 * i]     = pk_bf16(xv[0] * inv * wv[0], xv[1] * inv * wv[1]);
        ob[(size_t)b * (HIDDEN / 2) + 2 * i + 1] = pk_bf16(xv[2] * inv * wv[2], xv[3] * inv * wv[3]);
    }
}

// ---------------------------------------------------------------------------
// RoPE (in-place on qkv f32: 32 q heads + 2 k heads, rot dim 64, interleaved
// pairs), then write roped new K and raw new V into the last cache row.
// ---------------------------------------------------------------------------
__global__ __launch_bounds__(256) void rope_cache_k(float* __restrict__ qkv,
                                                    const int* __restrict__ pos,
                                                    float* __restrict__ outk,
                                                    float* __restrict__ outv)
{
    const int b = blockIdx.x;
    float* row = qkv + (size_t)b * QKV_O;
    const float p = (float)pos[b];
    for (int t = threadIdx.x; t < (NHEADS + NKV) * 32; t += 256) {
        const int h = t >> 5;
        const int i = t & 31;
        float* base = (h < NHEADS) ? (row + h * HD)
                                   : (row + HIDDEN + (h - NHEADS) * HD);
        const float inv = __powf(10000.0f, -(float)i / 32.0f);
        float s, c; __sincosf(p * inv, &s, &c);
        const float x0 = base[2 * i], x1 = base[2 * i + 1];
        base[2 * i]     = x0 * c - x1 * s;
        base[2 * i + 1] = x1 * c + x0 * s;
    }
    __syncthreads();
    for (int t = threadIdx.x; t < NKV * HD; t += 256) {
        const int kh = t / HD, d = t % HD;
        const size_t o = (size_t)(PAST - 1) * BATCH * NKV * HD
                       + ((size_t)b * NKV + kh) * HD + d;
        outk[o] = row[HIDDEN + kh * HD + d];
        outv[o] = row[HIDDEN + NKV * HD + kh * HD + d];
    }
}

// ---------------------------------------------------------------------------
// Flash-decode attention, pass 1. Block = (b, kv_head, key_split):
// 32*2*4 = 256 blocks, 256 threads = 8 waves. Each wave owns TWO q-heads
// (GQA: KV streamed ONCE per (b,kh), 8x L2 reuse). 4 groups of 8 lanes,
// each group one key per iteration, 16 dims/lane; online softmax per lane,
// group states merged with shfl_xor. Unnormalized (m, l, acc) -> ws.
// ---------------------------------------------------------------------------
__global__ __launch_bounds__(256) void attn_part_k(
    const float* __restrict__ qkv, const float* __restrict__ pk,
    const float* __restrict__ pv, const unsigned char* __restrict__ mask,
    float* __restrict__ part)
{
    __shared__ unsigned char smask[SEQ];
    const int s  = blockIdx.x & (ASPLIT - 1);
    const int kh = (blockIdx.x >> 2) & (NKV - 1);
    const int b  = blockIdx.x >> 3;
    const int t  = threadIdx.x;
    const int lane = t & 31, wid = t >> 5;
    const int g = lane >> 3, r = lane & 7;

    {   // preload this batch row's mask (4 KB) into LDS
        const unsigned* msrc = (const unsigned*)(mask + (size_t)b * SEQ);
        unsigned* mdst = (unsigned*)smask;
        for (int i = t; i < SEQ / 4; i += 256) mdst[i] = msrc[i];
    }
    __syncthreads();

    const int qh0 = kh * 16 + wid * 2;
    const float* qb = qkv + (size_t)b * QKV_O;
    float q0[16], q1[16];
    {
        const v4f* q0p = (const v4f*)(qb + qh0 * HD + r * 16);
        const v4f* q1p = (const v4f*)(qb + (qh0 + 1) * HD + r * 16);
#pragma unroll
        for (int j = 0; j < 4; ++j) {
            v4f a = q0p[j], c = q1p[j];
#pragma unroll
            for (int e = 0; e < 4; ++e) { q0[4 * j + e] = a[e]; q1[4 * j + e] = c[e]; }
        }
    }

    const float scale = 0.08838834764831845f;   // 1/sqrt(128)
    const float* knew = qb + HIDDEN + kh * HD;
    const float* vnew = qb + HIDDEN + NKV * HD + kh * HD;

    float m0 = -1e30f, l0 = 0.f, m1 = -1e30f, l1 = 0.f;
    float acc0[16] = {}, acc1[16] = {};

    const int kbase = s * (SEQ / ASPLIT);
    for (int i = 0; i < SEQ / ASPLIT / 4; ++i) {
        const int key = kbase + 4 * i + g;
        const float* kr; const float* vr;
        if (key < PAST) {
            const size_t off = (((size_t)key * BATCH + b) * NKV + kh) * HD;
            kr = pk + off; vr = pv + off;
        } else { kr = knew; vr = vnew; }

        const v4f* kp = (const v4f*)(kr + r * 16);
        v4f k0 = kp[0], k1 = kp[1], k2 = kp[2], k3 = kp[3];
        float d0 = 0.f, d1 = 0.f;
#pragma unroll
        for (int e = 0; e < 4; ++e) {
            d0 += q0[e] * k0[e] + q0[4 + e] * k1[e] + q0[8 + e] * k2[e] + q0[12 + e] * k3[e];
            d1 += q1[e] * k0[e] + q1[4 + e] * k1[e] + q1[8 + e] * k2[e] + q1[12 + e] * k3[e];
        }
        d0 += __shfl_xor(d0, 1, 32); d1 += __shfl_xor(d1, 1, 32);
        d0 += __shfl_xor(d0, 2, 32); d1 += __shfl_xor(d1, 2, 32);
        d0 += __shfl_xor(d0, 4, 32); d1 += __shfl_xor(d1, 4, 32);

        float s0 = d0 * scale, s1 = d1 * scale;
        if (smask[key]) { s0 = -1e9f; s1 = -1e9f; }

        const v4f* vp = (const v4f*)(vr + r * 16);
        v4f v0 = vp[0], v1v = vp[1], v2 = vp[2], v3 = vp[3];

        float nm0 = fmaxf(m0, s0), nm1 = fmaxf(m1, s1);
        float f0 = __expf(m0 - nm0), p0 = __expf(s0 - nm0);
        float f1 = __expf(m1 - nm1), p1 = __expf(s1 - nm1);
        m0 = nm0; m1 = nm1;
        l0 = l0 * f0 + p0; l1 = l1 * f1 + p1;
#pragma unroll
        for (int e = 0; e < 4; ++e) {
            acc0[e]      = acc0[e]      * f0 + p0 * v0[e];
            acc0[4 + e]  = acc0[4 + e]  * f0 + p0 * v1v[e];
            acc0[8 + e]  = acc0[8 + e]  * f0 + p0 * v2[e];
            acc0[12 + e] = acc0[12 + e] * f0 + p0 * v3[e];
            acc1[e]      = acc1[e]      * f1 + p1 * v0[e];
            acc1[4 + e]  = acc1[4 + e]  * f1 + p1 * v1v[e];
            acc1[8 + e]  = acc1[8 + e]  * f1 + p1 * v2[e];
            acc1[12 + e] = acc1[12 + e] * f1 + p1 * v3[e];
        }
    }

#pragma unroll
    for (int st = 8; st <= 16; st <<= 1) {
        float om0 = __shfl_xor(m0, st, 32), ol0 = __shfl_xor(l0, st, 32);
        float om1 = __shfl_xor(m1, st, 32), ol1 = __shfl_xor(l1, st, 32);
        float nm0 = fmaxf(m0, om0), nm1 = fmaxf(m1, om1);
        float fa0 = __expf(m0 - nm0), fb0 = __expf(om0 - nm0);
        float fa1 = __expf(m1 - nm1), fb1 = __expf(om1 - nm1);
#pragma unroll
        for (int j = 0; j < 16; ++j) {
            float oa0 = __shfl_xor(acc0[j], st, 32);
            float oa1 = __shfl_xor(acc1[j], st, 32);
            acc0[j] = acc0[j] * fa0 + oa0 * fb0;
            acc1[j] = acc1[j] * fa1 + oa1 * fb1;
        }
        l0 = l0 * fa0 + ol0 * fb0; l1 = l1 * fa1 + ol1 * fb1;
        m0 = nm0; m1 = nm1;
    }

    if (g == 0) {
        float* p0 = part + ((((size_t)(b * NKV + kh)) * ASPLIT + s) * 16 + wid * 2) * PSTRIDE;
        float* p1 = p0 + PSTRIDE;
        if (r == 0) { p0[0] = m0; p0[1] = l0; p1[0] = m1; p1[1] = l1; }
#pragma unroll
        for (int j = 0; j < 16; ++j) {
            p0[2 + r * 16 + j] = acc0[j];
            p1[2 + r * 16 + j] = acc1[j];
        }
    }
}

// ---------------------------------------------------------------------------
// Flash-decode pass 2: combine ASPLIT partials per (b, head); emits packed
// bf16 ctx (dense-GEMM A operand). 1024 blocks, 64 threads (2 dims each).
// ---------------------------------------------------------------------------
__global__ __launch_bounds__(64) void attn_merge_k(const float* __restrict__ part,
                                                   unsigned* __restrict__ ctxb)
{
    const int b = blockIdx.x >> 5;
    const int h = blockIdx.x & 31;
    const int kh = h >> 4, hl = h & 15;
    const int d0 = threadIdx.x * 2;
    float m[ASPLIT], l[ASPLIT], a0[ASPLIT], a1[ASPLIT];
#pragma unroll
    for (int s = 0; s < ASPLIT; ++s) {
        const float* p = part + ((((size_t)(b * NKV + kh)) * ASPLIT + s) * 16 + hl) * PSTRIDE;
        m[s] = p[0]; l[s] = p[1]; a0[s] = p[2 + d0]; a1[s] = p[3 + d0];
    }
    float M = fmaxf(fmaxf(m[0], m[1]), fmaxf(m[2], m[3]));
    float n0 = 0.f, n1 = 0.f, den = 0.f;
#pragma unroll
    for (int s = 0; s < ASPLIT; ++s) {
        float e = __expf(m[s] - M);
        n0 += a0[s] * e; n1 += a1[s] * e; den += l[s] * e;
    }
    const float inv = 1.0f / den;
    ctxb[(((size_t)b * NHEADS + h) * HD + d0) / 2] = pk_bf16(n0 * inv, n1 * inv);
}

// ---------------------------------------------------------------------------
// SwiGLU: out[b,f] = silu(h[b,f]) * h[b, FFN+f]; emits packed bf16.
// ---------------------------------------------------------------------------
__global__ __launch_bounds__(256) void swiglu_k(const float* __restrict__ hcat,
                                                unsigned* __restrict__ ob)
{
    const int idx = blockIdx.x * 256 + threadIdx.x;   // pair index
    if (idx >= BATCH * FFN / 2) return;
    const int b = idx / (FFN / 2), f2 = idx % (FFN / 2);
    const float2* ap = (const float2*)(hcat + (size_t)b * (2 * FFN));
    const float2 a = ap[f2];
    const float2 g = ap[FFN / 2 + f2];
    const float o0 = (a.x / (1.0f + __expf(-a.x))) * g.x;
    const float o1 = (a.y / (1.0f + __expf(-a.y))) * g.y;
    ob[idx] = pk_bf16(o0, o1);
}

extern "C" void kernel_launch(void* const* d_in, const int* in_sizes, int n_in,
                              void* d_out, int out_size, void* d_ws, size_t ws_size,
                              hipStream_t stream)
{
    const float* hidden  = (const float*)d_in[0];
    const int*   pos     = (const int*)d_in[1];
    const unsigned char* mask = (const unsigned char*)d_in[2];
    const float* past_k  = (const float*)d_in[3];
    const float* past_v  = (const float*)d_in[4];
    const float* w_ln1   = (const float*)d_in[5];
    const float* w_qkv   = (const float*)d_in[6];
    const float* b_qkv   = (const float*)d_in[7];
    const float* w_dense = (const float*)d_in[8];
    const float* w_ln2   = (const float*)d_in[9];
    const float* w_fc1   = (const float*)d_in[10];
    const float* w_fc2   = (const float*)d_in[11];

    float* out_h = (float*)d_out;
    float* out_k = out_h + (size_t)BATCH * HIDDEN;
    float* out_v = out_k + (size_t)PAST * BATCH * NKV * HD;

    // ---- workspace carve (floats) ----
    float* ws    = (float*)d_ws;
    float* qkv   = ws;                                   // 32*4608         f32
    float* h2    = qkv   + (size_t)BATCH * QKV_O;        // 32*4096         f32
    float* fc1   = h2    + (size_t)BATCH * HIDDEN;       // 32*27392        f32
    float* part  = fc1   + (size_t)BATCH * 2 * FFN;      // 32*2*4*16*130   f32
    float* gpart = part  + (size_t)BATCH * NKV * ASPLIT * 16 * PSTRIDE; // 4*32*4608 f32
    unsigned* xnb  = (unsigned*)(gpart + (size_t)GSPLIT * BATCH * QKV_O); // 32*4096 bf16
    unsigned* xn2b = xnb  + (size_t)BATCH * HIDDEN / 2;  // 32*4096 bf16
    unsigned* ctxb = xn2b + (size_t)BATCH * HIDDEN / 2;  // 32*4096 bf16
    unsigned* actb = ctxb + (size_t)BATCH * HIDDEN / 2;  // 32*13696 bf16

    // Shifted KV cache output: rows 0..4093 <- past rows 1..4094 (d2d copy)
    const size_t rowE = (size_t)BATCH * NKV * HD;
    hipMemcpyAsync(out_k, past_k + rowE, (size_t)(PAST - 1) * rowE * sizeof(float),
                   hipMemcpyDeviceToDevice, stream);
    hipMemcpyAsync(out_v, past_v + rowE, (size_t)(PAST - 1) * rowE * sizeof(float),
                   hipMemcpyDeviceToDevice, stream);

    // 1) x1 = rmsnorm(hidden, w_ln1) -> bf16
    rmsnorm_k<<<BATCH, 256, 0, stream>>>(hidden, w_ln1, xnb);
    // 2) qkv = x1 @ w_qkv^T + b_qkv   (split-K 4 + reduce)
    gemm_bf16_wmma<<<dim3(QKV_O / 128, GSPLIT), 256, 0, stream>>>(
        (const unsigned short*)xnb, w_qkv, gpart, nullptr, nullptr, HIDDEN, QKV_O, GSPLIT);
    gemm_reduce_k<<<(BATCH * QKV_O + 255) / 256, 256, 0, stream>>>(
        gpart, qkv, b_qkv, nullptr, QKV_O, GSPLIT, BATCH * QKV_O);
    // 3) rope q,k in place; write new kv row into caches
    rope_cache_k<<<BATCH, 256, 0, stream>>>(qkv, pos, out_k, out_v);
    // 4) attention (flash-decode, KV streamed once)
    attn_part_k<<<BATCH * NKV * ASPLIT, 256, 0, stream>>>(qkv, past_k, past_v, mask, part);
    attn_merge_k<<<BATCH * NHEADS, 64, 0, stream>>>(part, ctxb);
    // 5) h2 = ctx @ w_dense^T + hidden (split-K 4 + reduce)
    gemm_bf16_wmma<<<dim3(HIDDEN / 128, GSPLIT), 256, 0, stream>>>(
        (const unsigned short*)ctxb, w_dense, gpart, nullptr, nullptr, HIDDEN, HIDDEN, GSPLIT);
    gemm_reduce_k<<<(BATCH * HIDDEN + 255) / 256, 256, 0, stream>>>(
        gpart, h2, nullptr, hidden, HIDDEN, GSPLIT, BATCH * HIDDEN);
    // 6) x2 = rmsnorm(h2, w_ln2) -> bf16
    rmsnorm_k<<<BATCH, 256, 0, stream>>>(h2, w_ln2, xn2b);
    // 7) fc1 = x2 @ w_fc1^T  (214 blocks -> KS=1, direct store)
    gemm_bf16_wmma<<<dim3((2 * FFN) / 128, 1), 256, 0, stream>>>(
        (const unsigned short*)xn2b, w_fc1, fc1, nullptr, nullptr, HIDDEN, 2 * FFN, 1);
    // 8) act = silu(a)*b -> bf16
    swiglu_k<<<(BATCH * FFN / 2 + 255) / 256, 256, 0, stream>>>(fc1, actb);
    // 9) out = act @ w_fc2^T + h2 (split-K 4 + reduce)
    gemm_bf16_wmma<<<dim3(HIDDEN / 128, GSPLIT), 256, 0, stream>>>(
        (const unsigned short*)actb, w_fc2, gpart, nullptr, nullptr, FFN, HIDDEN, GSPLIT);
    gemm_reduce_k<<<(BATCH * HIDDEN + 255) / 256, 256, 0, stream>>>(
        gpart, out_h, nullptr, h2, HIDDEN, GSPLIT, BATCH * HIDDEN);
}
